// Model_82652350644349
// MI455X (gfx1250) — matrix operand order repaired and checked by reading, not compiled, not run on hardware
//
#include <hip/hip_runtime.h>

// Windowed attention w/ decomposed rel-pos (ViTDet/SAM): B=128, NH=16, S=196 (14x14), D=80.
// One workgroup per (batch, head); K/Q/V^T staged in LDS as f16; all matmuls via
// v_wmma_f32_16x16x32_f16; softmax + bias in f32 VALU.

#define SEQ   196
#define HH    14
#define HD    80
#define NH    16
#define CH    (NH*HD)      // 1280 channel stride in global q/k/v/out
#define STQ   104          // padded f16 channel stride for Q/K rows (>=96)
#define STV   224          // padded f16 t-stride for V^T rows (7*32)
#define STP   40           // P scratch row stride
#define NT    13           // 13 tiles of 16 rows cover 208 >= 196

typedef __attribute__((ext_vector_type(16))) _Float16 v16h;
typedef __attribute__((ext_vector_type(8)))  _Float16 v8h;
typedef __attribute__((ext_vector_type(8)))  float    v8f;
typedef __attribute__((ext_vector_type(4)))  float    v4f;

__device__ __forceinline__ v8f zero8() {
  v8f z;
#pragma unroll
  for (int i = 0; i < 8; ++i) z[i] = 0.0f;
  return z;
}

// A-matrix (16x32 f16) lane layout: lanes 0-15 = rows, K {c0..c0+7, c0+16..c0+23};
// lanes 16-31 same rows, K {c0+8..15, c0+24..31}.
__device__ __forceinline__ v16h load_a16(const _Float16* base, int row0, int stride,
                                         int c0, int ln, int hi) {
  const _Float16* p = base + (row0 + ln) * stride + c0 + hi * 8;
  v8h lo = *(const v8h*)p;
  v8h hh = *(const v8h*)(p + 16);
  v16h r;
#pragma unroll
  for (int i = 0; i < 8; ++i) { r[i] = lo[i]; r[i + 8] = hh[i]; }
  return r;
}

// B-matrix (32x16 f16) lane layout: lane = column, lanes 0-15 hold K {c0..c0+7, c0+16..c0+23},
// lanes 16-31 hold K {c0+8..15, c0+24..31}.
__device__ __forceinline__ v16h load_b16(const _Float16* base, int col0, int stride,
                                         int c0, int ln, int hi) {
  const _Float16* p = base + (col0 + ln) * stride + c0 + hi * 8;
  v8h lo = *(const v8h*)p;
  v8h hh = *(const v8h*)(p + 16);
  v16h r;
#pragma unroll
  for (int i = 0; i < 8; ++i) { r[i] = lo[i]; r[i + 8] = hh[i]; }
  return r;
}

__global__ __launch_bounds__(256) void win_attn_relpos(
    const float* __restrict__ q, const float* __restrict__ k,
    const float* __restrict__ v, const float* __restrict__ rph,
    const float* __restrict__ rpw, float* __restrict__ out) {
  __shared__ __align__(16) _Float16 sQ[208 * STQ];   // scaled q, f16
  __shared__ __align__(16) _Float16 sK[208 * STQ];   // k, f16
  __shared__ __align__(16) _Float16 sVt[HD * STV];   // v transposed [d][t], f16
  __shared__ float sRh[SEQ * HH];                    // rel_h bias [s][k]
  __shared__ float sRw[SEQ * HH];                    // rel_w bias [s][l]
  __shared__ __align__(16) _Float16 sP[8][16 * STP]; // per-wave P transpose scratch

  const int tid  = threadIdx.x;
  const int head = blockIdx.x;
  const int b    = blockIdx.y;
  const size_t base = (size_t)b * SEQ * CH + (size_t)head * HD;
  const float* qb = q + base;
  const float* kb = k + base;
  const float* vb = v + base;
  float* ob = out + base;

  // ---- zero padded LDS regions (pad rows/cols must be 0 for WMMA) ----
  {
    uint32_t* z = (uint32_t*)sQ;
#pragma unroll 1
    for (int i = tid; i < 208 * STQ / 2; i += 256) z[i] = 0u;
  }
  {
    uint32_t* z = (uint32_t*)sK;
#pragma unroll 1
    for (int i = tid; i < 208 * STQ / 2; i += 256) z[i] = 0u;
  }
  {
    uint32_t* z = (uint32_t*)sVt;
#pragma unroll 1
    for (int i = tid; i < HD * STV / 2; i += 256) z[i] = 0u;
  }
  __syncthreads();

  const float scale = 0.11180339887498949f;  // 80^-0.5 folded into Q

  // ---- stage Q (scaled) and K as f16 rows ----
#pragma unroll 1
  for (int i = tid; i < SEQ * (HD / 4); i += 256) {
    int s = i / (HD / 4);
    int c = (i % (HD / 4)) * 4;
    v4f qv = *(const v4f*)(qb + (size_t)s * CH + c);
    v4f kv = *(const v4f*)(kb + (size_t)s * CH + c);
    _Float16* dq = sQ + s * STQ + c;
    _Float16* dk = sK + s * STQ + c;
#pragma unroll
    for (int j = 0; j < 4; ++j) {
      dq[j] = (_Float16)(qv[j] * scale);
      dk[j] = (_Float16)kv[j];
    }
  }
  // ---- stage V transposed: sVt[d][t] ----
#pragma unroll 1
  for (int i = tid; i < SEQ * HD; i += 256) {
    int t = i / HD, d = i % HD;
    sVt[d * STV + t] = (_Float16)vb[(size_t)t * CH + d];
  }
  // ---- rel-pos bias tables: rel_h[s][k] = q[s,:]·Rh[h(s)-k+13,:], same for w ----
#pragma unroll 1
  for (int i = tid; i < SEQ * HH; i += 256) {
    int s = i / HH, kk = i % HH;
    int h = s / HH, w = s % HH;
    const float* qrow = qb + (size_t)s * CH;
    const float* rh = rph + (h - kk + HH - 1) * HD;
    const float* rw = rpw + (w - kk + HH - 1) * HD;
    float ah = 0.0f, aw = 0.0f;
    for (int c = 0; c < HD; ++c) {
      float qc = qrow[c];
      ah += qc * rh[c];
      aw += qc * rw[c];
    }
    sRh[i] = ah;
    sRw[i] = aw;
  }
  __syncthreads();

  const int lane = tid & 31;
  const int ln   = lane & 15;   // A-row / B-col / C-col index
  const int hi   = lane >> 4;   // half-wave select
  const int wave = tid >> 5;

#pragma unroll 1
  for (int mt = wave; mt < NT; mt += 8) {
    // ---- sim strip: 16 query rows x 208 key cols, f32 accumulators ----
    v8f acc[NT];
#pragma unroll
    for (int i = 0; i < NT; ++i) acc[i] = zero8();
#pragma unroll
    for (int kc = 0; kc < 3; ++kc) {
      v16h a = load_a16(sQ, mt * 16, STQ, kc * 32, ln, hi);
#pragma unroll
      for (int nt = 0; nt < NT; ++nt) {
        v16h bm = load_b16(sK, nt * 16, STQ, kc * 32, ln, hi);
        acc[nt] = __builtin_amdgcn_wmma_f32_16x16x32_f16(
            false, a, false, bm, (short)0, acc[nt], false, false);
      }
    }
    // ---- add decomposed rel-pos bias; mask pad columns ----
#pragma unroll
    for (int nt = 0; nt < NT; ++nt) {
      int t = nt * 16 + ln;
      if (t < SEQ) {
        int kq = t / HH, lq = t % HH;
#pragma unroll
        for (int e = 0; e < 8; ++e) {
          int s = mt * 16 + e + 8 * hi;
          if (s < SEQ) acc[nt][e] += sRh[s * HH + kq] + sRw[s * HH + lq];
        }
      } else {
#pragma unroll
        for (int e = 0; e < 8; ++e) acc[nt][e] = -1e30f;
      }
    }
    // ---- softmax over the full row (row lives in one 16-lane half) ----
#pragma unroll
    for (int e = 0; e < 8; ++e) {
      float m = -1e30f;
#pragma unroll
      for (int nt = 0; nt < NT; ++nt) m = fmaxf(m, acc[nt][e]);
#pragma unroll
      for (int off = 1; off < 16; off <<= 1) m = fmaxf(m, __shfl_xor(m, off, 32));
      float sum = 0.0f;
#pragma unroll
      for (int nt = 0; nt < NT; ++nt) {
        float p = __expf(acc[nt][e] - m);
        acc[nt][e] = p;
        sum += p;
      }
#pragma unroll
      for (int off = 1; off < 16; off <<= 1) sum += __shfl_xor(sum, off, 32);
      float inv = 1.0f / sum;
#pragma unroll
      for (int nt = 0; nt < NT; ++nt) acc[nt][e] *= inv;
    }
    // ---- P @ V: re-lay P C->A via per-wave LDS scratch, chunk by chunk ----
    v8f o[5];
#pragma unroll
    for (int i = 0; i < 5; ++i) o[i] = zero8();
    _Float16* pw = sP[wave];
#pragma unroll
    for (int kc = 0; kc < 7; ++kc) {
#pragma unroll
      for (int hf = 0; hf < 2; ++hf) {
        int nt = 2 * kc + hf;
#pragma unroll
        for (int e = 0; e < 8; ++e) {
          float pv = (nt < NT) ? acc[nt][e] : 0.0f;
          pw[(e + 8 * hi) * STP + hf * 16 + ln] = (_Float16)pv;
        }
      }
      asm volatile("s_wait_dscnt 0" ::: "memory");
      v16h a = load_a16(pw, 0, STP, 0, ln, hi);
#pragma unroll
      for (int nd = 0; nd < 5; ++nd) {
        v16h bm = load_b16(sVt, nd * 16, STV, kc * 32, ln, hi);
        o[nd] = __builtin_amdgcn_wmma_f32_16x16x32_f16(
            false, a, false, bm, (short)0, o[nd], false, false);
      }
    }
    // ---- store (b, s, head*80 + d) in f32 ----
#pragma unroll
    for (int nd = 0; nd < 5; ++nd) {
      int d = nd * 16 + ln;
#pragma unroll
      for (int e = 0; e < 8; ++e) {
        int s = mt * 16 + e + 8 * hi;
        if (s < SEQ) ob[(size_t)s * CH + d] = o[nd][e];
      }
    }
  }
}

extern "C" void kernel_launch(void* const* d_in, const int* in_sizes, int n_in,
                              void* d_out, int out_size, void* d_ws, size_t ws_size,
                              hipStream_t stream) {
  const float* q   = (const float*)d_in[0];
  const float* k   = (const float*)d_in[1];
  const float* v   = (const float*)d_in[2];
  const float* rph = (const float*)d_in[3];
  const float* rpw = (const float*)d_in[4];
  float* out = (float*)d_out;
  int batches = in_sizes[0] / (SEQ * CH);  // 128
  dim3 grid(NH, batches);
  win_attn_relpos<<<grid, dim3(256), 0, stream>>>(q, k, v, rph, rpw, out);
}
